// AttnRes_46462956208774
// MI455X (gfx1250) — matrix operand order repaired and checked by reading, not compiled, not run on hardware
//
#include <hip/hip_runtime.h>

// Problem constants (match reference)
#define Bc  8
#define Tc  1024
#define Dc  768
#define Lc  8
#define Hc  12
#define HDc 64
#define BTc (Bc*Tc)

typedef __bf16 bf16_t;
typedef __attribute__((ext_vector_type(16))) __bf16 v16bf;
typedef __attribute__((ext_vector_type(8)))  float  v8f;

// ---------------------------------------------------------------------------
// WMMA helpers (CDNA5: D = A(16x32 bf16) * B(32x16 bf16) + C(16x16 f32))
// ---------------------------------------------------------------------------
__device__ __forceinline__ v8f wmma_bf16(v16bf a, v16bf b, v8f c) {
  return __builtin_amdgcn_wmma_f32_16x16x32_bf16(
      /*neg_a=*/false, a, /*neg_b=*/false, b,
      /*c_mod=*/(short)0, c, /*reuse_a=*/false, /*reuse_b=*/false);
}

// A-matrix 16x32 bf16 fragment. ISA layout:
//   lanes 0-15 : row M=lane, K 0..7 (v0-3) and K 16..23 (v4-7)
//   lanes 16-31: row M=lane-16, K 8..15 and K 24..31
// strideElems*2 and base must be 16B aligned.
__device__ __forceinline__ v16bf ld_afrag(const bf16_t* base, int strideElems) {
  int lane = threadIdx.x & 31;
  int half = lane >> 4, r = lane & 15;
  const bf16_t* p = base + (size_t)r * strideElems + half * 8;
  union { v16bf v; float4 f[2]; } u;
  u.f[0] = *(const float4*)p;
  u.f[1] = *(const float4*)(p + 16);
  return u.v;
}

// B-matrix 32x16 bf16 fragment, source stored as [n][k] (k contiguous):
//   lanes 0-15 : col N=lane, K 0..7 and K 16..23 ; lanes 16-31: col N=lane-16, K 8..15 and K 24..31
__device__ __forceinline__ v16bf ld_bfrag(const bf16_t* base, int strideElems) {
  int lane = threadIdx.x & 31;
  int half = lane >> 4, n = lane & 15;
  const bf16_t* p = base + (size_t)n * strideElems + half * 8;
  union { v16bf v; float4 f[2]; } u;
  u.f[0] = *(const float4*)p;
  u.f[1] = *(const float4*)(p + 16);
  return u.v;
}

__device__ __forceinline__ float gelu_tanh(float x) {
  float x3 = x * x * x;
  return 0.5f * x * (1.0f + tanhf(0.7978845608028654f * (x + 0.044715f * x3)));
}

// ---------------------------------------------------------------------------
// Tensor Data Mover: async DMA of a 2D bf16 tile (global -> LDS) with LDS
// row padding injected by the TDM itself (D# pad_interval / pad_amount).
// Descriptor bitfields per CDNA5 ISA.
// Guarded: falls back to manual staging when the builtin is absent.
// ---------------------------------------------------------------------------
#if __has_builtin(__builtin_amdgcn_tensor_load_to_lds)
#define HAVE_TDM 1
typedef unsigned int u32x4 __attribute__((ext_vector_type(4)));
typedef int          i32x4 __attribute__((ext_vector_type(4)));
typedef int          i32x8 __attribute__((ext_vector_type(8)));

__device__ __forceinline__ void tdm_load_2d_bf16(
    const bf16_t* gsrc, void* lds_dst,
    unsigned tensor_d0, unsigned tensor_d1,   // remaining tensor extent (elems)
    unsigned tile_d0, unsigned tile_d1,       // tile size (elems)
    unsigned stride_d0,                       // row stride (elems)
    unsigned pad_interval_code,               // pad every 2<<code DWORDs
    unsigned pad_amount_code) {               // pad (code+1) DWORDs
  unsigned long long ga = (unsigned long long)(size_t)gsrc;
  unsigned lds = (unsigned)(size_t)lds_dst;   // low 32 bits = LDS byte offset
  u32x4 g0;
  g0[0] = 1u;                                         // count=1, user D#
  g0[1] = lds;                                        // lds_addr
  g0[2] = (unsigned)(ga & 0xffffffffu);               // global_addr[31:0]
  g0[3] = (unsigned)((ga >> 32) & 0x1ffffffu)         // global_addr[56:32]
        | (2u << 30);                                 // type = 2 ("image")
  i32x8 g1;
  g1[0] = (int)((1u << 16)                            // data_size = 2 bytes
        | (1u << 20)                                  // pad_enable
        | (pad_interval_code << 22)
        | (pad_amount_code << 25));
  g1[1] = (int)((tensor_d0 & 0xffffu) << 16);         // tensor_dim0[15:0]
  g1[2] = (int)(((tensor_d0 >> 16) & 0xffffu)         // tensor_dim0[31:16]
        | ((tensor_d1 & 0xffffu) << 16));             // tensor_dim1[15:0]
  g1[3] = (int)(((tensor_d1 >> 16) & 0xffffu)         // tensor_dim1[31:16]
        | ((tile_d0 & 0xffffu) << 16));               // tile_dim0
  g1[4] = (int)(tile_d1 & 0xffffu);                   // tile_dim1 (tile_dim2=0)
  g1[5] = (int)stride_d0;                             // tensor_dim0_stride[31:0]
  g1[6] = 0;
  g1[7] = 0;
  i32x4 z4 = (i32x4)0;
#if defined(__clang_major__) && (__clang_major__ >= 23)
  i32x8 z8 = (i32x8)0;
  __builtin_amdgcn_tensor_load_to_lds(g0, g1, z4, z4, z8, 0);
#else
  __builtin_amdgcn_tensor_load_to_lds(g0, g1, z4, z4, 0);
#endif
}

__device__ __forceinline__ void tdm_wait0() {
#if __has_builtin(__builtin_amdgcn_s_wait_tensorcnt)
  __builtin_amdgcn_s_wait_tensorcnt(0);
#else
  asm volatile("s_wait_tensorcnt 0x0" ::: "memory");
#endif
}
#else
#define HAVE_TDM 0
#endif

// ---------------------------------------------------------------------------
// f32 -> bf16 pack kernel (4 elems / thread)
// ---------------------------------------------------------------------------
__global__ void f32_to_bf16_kernel(const float* __restrict__ in,
                                   bf16_t* __restrict__ out, long n4) {
  long i = (long)blockIdx.x * blockDim.x + threadIdx.x;
  if (i >= n4) return;
  float4 f = ((const float4*)in)[i];
  union { bf16_t h[4]; uint2 u; } r;
  r.h[0] = (bf16_t)f.x; r.h[1] = (bf16_t)f.y;
  r.h[2] = (bf16_t)f.z; r.h[3] = (bf16_t)f.w;
  ((uint2*)out)[i] = r.u;
}

// ---------------------------------------------------------------------------
// Fused depth-aggregate + rms_norm*s -> bf16 (one block per (b,t) row)
// ---------------------------------------------------------------------------
struct VPtrs { const float* p[6]; };

__device__ __forceinline__ float block_reduce_sum(float v, float* sbuf) {
#pragma unroll
  for (int m = 16; m >= 1; m >>= 1) v += __shfl_xor(v, m, 32);
  int wid = threadIdx.x >> 5, lid = threadIdx.x & 31;
  if (lid == 0) sbuf[wid] = v;
  __syncthreads();
  if (wid == 0) {
    float t = (lid < 8) ? sbuf[lid] : 0.0f;
#pragma unroll
    for (int m = 4; m >= 1; m >>= 1) t += __shfl_xor(t, m, 32);
    if (lid == 0) sbuf[0] = t;
  }
  __syncthreads();
  float r = sbuf[0];
  __syncthreads();
  return r;
}

__global__ __launch_bounds__(256) void aggregate_rms_kernel(
    VPtrs V, int n, const float* __restrict__ qvec,
    const float* __restrict__ lns, int lidx, bf16_t* __restrict__ out) {
  __shared__ float sbuf[8];
  const float eps = 1.1920929e-07f;  // finfo(f32).eps
  int row = blockIdx.x, tid = threadIdx.x;
  float qv[3], vreg[6][3], ssq[6], dot[6];
#pragma unroll
  for (int k = 0; k < 3; k++) qv[k] = qvec[tid + k * 256];
#pragma unroll
  for (int j = 0; j < 6; j++) {
    ssq[j] = 0.f; dot[j] = 0.f;
    vreg[j][0] = vreg[j][1] = vreg[j][2] = 0.f;
    if (j < n) {
      const float* vr = V.p[j] + (size_t)row * Dc;
#pragma unroll
      for (int k = 0; k < 3; k++) {
        float v = vr[tid + k * 256];
        vreg[j][k] = v; ssq[j] += v * v; dot[j] += qv[k] * v;
      }
    }
  }
  float logit[6];
  for (int j = 0; j < 6; j++) {
    if (j < n) {  // n is block-uniform -> barriers safe
      float sq = block_reduce_sum(ssq[j], sbuf);
      float dt = block_reduce_sum(dot[j], sbuf);
      logit[j] = dt * rsqrtf(sq * (1.0f / Dc) + eps);
    } else logit[j] = -1e30f;
  }
  float mx = -1e30f;
#pragma unroll
  for (int j = 0; j < 6; j++) mx = fmaxf(mx, logit[j]);
  float w[6], wsum = 0.f;
#pragma unroll
  for (int j = 0; j < 6; j++) { w[j] = (j < n) ? __expf(logit[j] - mx) : 0.f; wsum += w[j]; }
  float inv = 1.0f / wsum;
  float h[3] = {0.f, 0.f, 0.f};
#pragma unroll
  for (int j = 0; j < 6; j++)
#pragma unroll
    for (int k = 0; k < 3; k++) h[k] += (w[j] * inv) * vreg[j][k];
  float sh = h[0]*h[0] + h[1]*h[1] + h[2]*h[2];
  sh = block_reduce_sum(sh, sbuf);
  float scale = rsqrtf(sh * (1.0f / Dc) + eps) * lns[lidx];
#pragma unroll
  for (int k = 0; k < 3; k++)
    out[(size_t)row * Dc + tid + k * 256] = (bf16_t)(h[k] * scale);
}

// ---------------------------------------------------------------------------
// Tiled bf16 WMMA GEMM: C[M,N] = A[M,K] * B[K,N]
//  block 256 thr = 8 waves; block tile 128x128, BK=32; wave tile 32x64
//  A tile staged by the Tensor Data Mover (TDM injects the LDS row padding);
//  B tile staged manually (needs a transpose the TDM cannot do).
//  MODE 0: C f32    MODE 1: Cb = bf16(gelu(acc))    MODE 2: C = resid + colscale*acc
// ---------------------------------------------------------------------------
#define LDSK 48  // 32 data + 16 pad elems: keeps b128 LDS loads 16B aligned

template <int MODE>
__global__ __launch_bounds__(256) void gemm_bf16_kernel(
    const bf16_t* __restrict__ A, const bf16_t* __restrict__ B,
    float* __restrict__ C, bf16_t* __restrict__ Cb,
    const float* __restrict__ resid, const float* __restrict__ colscale,
    int M, int N, int K) {
  __shared__ __align__(16) bf16_t As[128 * LDSK];
  __shared__ __align__(16) bf16_t Bs[128 * LDSK];  // stored transposed: [n][k]
  int tid = threadIdx.x, lane = tid & 31, w = tid >> 5;
  int m0 = blockIdx.y * 128, n0 = blockIdx.x * 128;
  int wm = (w >> 1) * 32, wn = (w & 1) * 64;

  v8f acc[2][4] = {};
  int nkb = K >> 5;
  for (int kb = 0; kb < nkb; ++kb) {
    int k0 = kb << 5;
    __syncthreads();
    // ---- stage A 128x32 (row-major) ----
#if HAVE_TDM
    if (w == 0) {  // wave 0 issues the DMA; TDM ignores EXEC
      // row = 32 elems = 16 DWORDs -> pad_interval code 3 (2<<3 = 16 DWORDs)
      // pad 16 elems = 8 DWORDs     -> pad_amount  code 7 (7+1 = 8 DWORDs)
      tdm_load_2d_bf16(A + (size_t)(m0)*K + k0, As,
                       (unsigned)(K - k0), (unsigned)(M - m0),
                       32u, 128u, (unsigned)K, 3u, 7u);
      tdm_wait0();
    }
#else
#pragma unroll
    for (int i = 0; i < 4; i++) {
      int c = tid + i * 256;         // 1024 8-byte chunks
      int r = c >> 3, kc = c & 7;
      uint2 d = *(const uint2*)(A + (size_t)(m0 + r) * K + k0 + kc * 4);
      *(uint2*)(&As[r * LDSK + kc * 4]) = d;
    }
#endif
    // ---- stage B 32x128 -> transposed LDS [n][k] ----
#pragma unroll
    for (int i = 0; i < 4; i++) {
      int c = tid + i * 256;
      int kk = c >> 5, nc = c & 31;
      union { uint2 u; bf16_t h[4]; } u2;
      u2.u = *(const uint2*)(B + (size_t)(k0 + kk) * N + n0 + nc * 4);
#pragma unroll
      for (int e = 0; e < 4; e++) Bs[(nc * 4 + e) * LDSK + kk] = u2.h[e];
    }
    if (kb + 1 < nkb) {  // gfx1250 global_prefetch_b8 for next K tile
      __builtin_prefetch(A + (size_t)(m0 + (tid & 127)) * K + k0 + 32);
      __builtin_prefetch(B + (size_t)(k0 + 32 + (tid & 31)) * N + n0 + ((tid >> 5) & 3) * 32);
    }
    __syncthreads();

    v16bf a0 = ld_afrag(&As[(wm +  0) * LDSK], LDSK);
    v16bf a1 = ld_afrag(&As[(wm + 16) * LDSK], LDSK);
#pragma unroll
    for (int j = 0; j < 4; j++) {
      v16bf bj = ld_bfrag(&Bs[(wn + j * 16) * LDSK], LDSK);
      acc[0][j] = wmma_bf16(a0, bj, acc[0][j]);
      acc[1][j] = wmma_bf16(a1, bj, acc[1][j]);
    }
  }

  // epilogue: C layout -> lane 0-15: M=r, N=lane ; lane 16-31: M=r+8, N=lane-16
  int half = lane >> 4, l15 = lane & 15;
#pragma unroll
  for (int i = 0; i < 2; i++)
#pragma unroll
    for (int j = 0; j < 4; j++)
#pragma unroll
      for (int r = 0; r < 8; r++) {
        int row = m0 + wm + i * 16 + r + 8 * half;
        int col = n0 + wn + j * 16 + l15;
        size_t idx = (size_t)row * N + col;
        float v = acc[i][j][r];
        if (MODE == 0) {
          C[idx] = v;
        } else if (MODE == 1) {
          Cb[idx] = (bf16_t)gelu_tanh(v);
        } else {
          float rv = resid ? resid[idx] : 0.0f;
          C[idx] = rv + colscale[col] * v;
        }
      }
}

// ---------------------------------------------------------------------------
// Flash attention (causal MHA), one wave per (b, h, 16-query-row tile)
//  qkv : bf16 [B*T, 3D]  (q | k | v, heads = h*64+hd inside each D slice)
//  ctx : bf16 [B*T, D]
// ---------------------------------------------------------------------------
__global__ __launch_bounds__(32) void flash_attn_kernel(
    const bf16_t* __restrict__ qkv, bf16_t* __restrict__ ctx) {
  __shared__ __align__(16) bf16_t pl[16 * 32];  // P tile, row-major
  __shared__ __align__(16) bf16_t vt[64 * 32];  // V^T tile: [hd][key]
  const int ld = 3 * Dc;
  int lane = threadIdx.x & 31;
  int half = lane >> 4, l15 = lane & 15;
  int qt0 = blockIdx.x * 16;
  int b = blockIdx.y / Hc, h = blockIdx.y % Hc;
  const bf16_t* qbase = qkv + (size_t)(b * Tc + qt0) * ld + h * HDc;
  const bf16_t* kbase = qkv + (size_t)(b * Tc) * ld + Dc + h * HDc;
  const bf16_t* vbase = qkv + (size_t)(b * Tc) * ld + 2 * Dc + h * HDc;

  v16bf qa0 = ld_afrag(qbase, ld);        // Q rows, hd 0..31
  v16bf qa1 = ld_afrag(qbase + 32, ld);   // Q rows, hd 32..63

  v8f o0 = {}, o1 = {}, o2 = {}, o3 = {};
  float mrow[8], lrow[8];
#pragma unroll
  for (int j = 0; j < 8; j++) { mrow[j] = -1e30f; lrow[j] = 0.f; }

  int lastkb = (qt0 + 15) >> 5;
  for (int kb2 = 0; kb2 <= lastkb; ++kb2) {
    int k0 = kb2 * 32;
    // S = Q * K^T  (B fragment = K rows, hd contiguous in memory)
    v16bf kb00 = ld_bfrag(kbase + (size_t)k0 * ld, ld);
    v16bf kb01 = ld_bfrag(kbase + (size_t)k0 * ld + 32, ld);
    v16bf kb10 = ld_bfrag(kbase + (size_t)(k0 + 16) * ld, ld);
    v16bf kb11 = ld_bfrag(kbase + (size_t)(k0 + 16) * ld + 32, ld);
    v8f s0 = {}, s1 = {};
    s0 = wmma_bf16(qa0, kb00, s0);
    s0 = wmma_bf16(qa1, kb01, s0);
    s1 = wmma_bf16(qa0, kb10, s1);
    s1 = wmma_bf16(qa1, kb11, s1);

    // stage V tile transposed into LDS: vt[hd][key_local]
    {
      const bf16_t* vp = vbase + (size_t)(k0 + lane) * ld;
#pragma unroll
      for (int c = 0; c < 8; c++) {
        union { uint2 u; bf16_t h4[4]; } wv;
        wv.u = *(const uint2*)(vp + c * 4);
#pragma unroll
        for (int e = 0; e < 4; e++) vt[(c * 4 + e) * 32 + lane] = wv.h4[e];
      }
    }

    // online softmax (rows live across 16 lanes of a half-wave)
#pragma unroll
    for (int j = 0; j < 8; j++) {
      int rowg = qt0 + j + 8 * half;
      float e0 = s0[j] * 0.125f;  // 1/sqrt(64)
      float e1 = s1[j] * 0.125f;
      if (k0 + l15 > rowg)      e0 = -1e30f;
      if (k0 + 16 + l15 > rowg) e1 = -1e30f;
      float rm = fmaxf(e0, e1);
      rm = fmaxf(rm, __shfl_xor(rm, 8, 32));
      rm = fmaxf(rm, __shfl_xor(rm, 4, 32));
      rm = fmaxf(rm, __shfl_xor(rm, 2, 32));
      rm = fmaxf(rm, __shfl_xor(rm, 1, 32));
      float mnew = fmaxf(mrow[j], rm);
      float sc = __expf(mrow[j] - mnew);
      float p0 = __expf(e0 - mnew), p1 = __expf(e1 - mnew);
      float rs = p0 + p1;
      rs += __shfl_xor(rs, 8, 32);
      rs += __shfl_xor(rs, 4, 32);
      rs += __shfl_xor(rs, 2, 32);
      rs += __shfl_xor(rs, 1, 32);
      lrow[j] = lrow[j] * sc + rs;
      mrow[j] = mnew;
      o0[j] *= sc; o1[j] *= sc; o2[j] *= sc; o3[j] *= sc;
      int rl = j + 8 * half;
      pl[rl * 32 + l15]      = (bf16_t)p0;
      pl[rl * 32 + 16 + l15] = (bf16_t)p1;
    }

    // O += P(16x32) * V(32x64) ; same-wave LDS ops are in-order
    v16bf pa = ld_afrag(pl, 32);
    o0 = wmma_bf16(pa, ld_bfrag(vt +  0 * 32, 32), o0);
    o1 = wmma_bf16(pa, ld_bfrag(vt + 16 * 32, 32), o1);
    o2 = wmma_bf16(pa, ld_bfrag(vt + 32 * 32, 32), o2);
    o3 = wmma_bf16(pa, ld_bfrag(vt + 48 * 32, 32), o3);
  }

#pragma unroll
  for (int j = 0; j < 8; j++) {
    float inv = 1.0f / lrow[j];
    int rowg = qt0 + j + 8 * half;
    bf16_t* crow = ctx + (size_t)(b * Tc + rowg) * Dc + h * HDc;
    crow[ 0 + l15] = (bf16_t)(o0[j] * inv);
    crow[16 + l15] = (bf16_t)(o1[j] * inv);
    crow[32 + l15] = (bf16_t)(o2[j] * inv);
    crow[48 + l15] = (bf16_t)(o3[j] * inv);
  }
}

// ---------------------------------------------------------------------------
// Host orchestration
// ---------------------------------------------------------------------------
extern "C" void kernel_launch(void* const* d_in, const int* in_sizes, int n_in,
                              void* d_out, int out_size, void* d_ws, size_t ws_size,
                              hipStream_t stream) {
  (void)in_sizes; (void)n_in; (void)out_size; (void)ws_size;
  const float* xin  = (const float*)d_in[0];
  const float* apw  = (const float*)d_in[1];
  const float* mpw  = (const float*)d_in[2];
  const float* Wqkv = (const float*)d_in[3];
  const float* Wo   = (const float*)d_in[4];
  const float* Wm1  = (const float*)d_in[5];
  const float* Wm2  = (const float*)d_in[6];
  const float* ascl = (const float*)d_in[7];
  const float* mscl = (const float*)d_in[8];
  const float* lns  = (const float*)d_in[9];
  float* outp = (float*)d_out;

  char* ws = (char*)d_ws;
  size_t off = 0;
  auto alloc = [&](size_t bytes) -> void* {
    void* p = ws + off;
    off += (bytes + 255) & ~(size_t)255;
    return p;
  };
  const size_t BTD = (size_t)BTc * Dc;
  float* pool[5];
  for (int i = 0; i < 5; i++) pool[i] = (float*)alloc(BTD * 4);
  float*  qkvf = (float*) alloc((size_t)BTc * 3 * Dc * 4);
  bf16_t* qkvh = (bf16_t*)alloc((size_t)BTc * 3 * Dc * 2);
  bf16_t* ctxh = (bf16_t*)alloc(BTD * 2);
  bf16_t* xn1  = (bf16_t*)alloc(BTD * 2);
  bf16_t* xn2  = (bf16_t*)alloc(BTD * 2);
  bf16_t* m1h  = (bf16_t*)alloc((size_t)BTc * 4 * Dc * 2);
  size_t nWqkv = (size_t)Lc * Dc * 3 * Dc;
  size_t nWo   = (size_t)Lc * Dc * Dc;
  size_t nWm1  = (size_t)Lc * Dc * 4 * Dc;
  size_t nWm2  = nWm1;
  bf16_t* wqkvh = (bf16_t*)alloc(nWqkv * 2);
  bf16_t* woh   = (bf16_t*)alloc(nWo * 2);
  bf16_t* wm1h  = (bf16_t*)alloc(nWm1 * 2);
  bf16_t* wm2h  = (bf16_t*)alloc(nWm2 * 2);

  auto cvt = [&](const float* src, bf16_t* dst, size_t n) {
    long n4 = (long)(n / 4);
    int grid = (int)((n4 + 255) / 256);
    f32_to_bf16_kernel<<<grid, 256, 0, stream>>>(src, dst, n4);
  };
  cvt(Wqkv, wqkvh, nWqkv);
  cvt(Wo,   woh,   nWo);
  cvt(Wm1,  wm1h,  nWm1);
  cvt(Wm2,  wm2h,  nWm2);

  const float* hist[6]; int nh = 1; hist[0] = xin;
  const float* partial = xin;
  bool pzero = false;
  bool frozen[5] = {false, false, false, false, false};
  auto pick = [&](const float* cur) -> float* {
    for (int i = 0; i < 5; i++)
      if (!frozen[i] && pool[i] != cur) return pool[i];
    return pool[0];
  };
  auto freeze = [&](const float* p) {
    for (int i = 0; i < 5; i++) if (pool[i] == p) frozen[i] = true;
  };

  for (int l = 0; l < Lc; l++) {
    // --- aggregate over depth stack, fused rms_norm * ln_s -> bf16 ---
    VPtrs V{};
    for (int j = 0; j < nh; j++) V.p[j] = hist[j];
    V.p[nh] = partial;
    aggregate_rms_kernel<<<BTc, 256, 0, stream>>>(
        V, nh + 1, apw + (size_t)l * Dc, lns, l, xn1);

    if ((l & 1) == 0) {  // boundary: commit partial to history, zero partial
      hist[nh++] = partial;
      freeze(partial);
      pzero = true;
    }

    // --- QKV projection ---
    gemm_bf16_kernel<0><<<dim3(3 * Dc / 128, BTc / 128), 256, 0, stream>>>(
        xn1, wqkvh + (size_t)l * Dc * 3 * Dc, qkvf, nullptr, nullptr, nullptr,
        BTc, 3 * Dc, Dc);
    cvt(qkvf, qkvh, (size_t)BTc * 3 * Dc);

    // --- causal MHA ---
    flash_attn_kernel<<<dim3(Tc / 16, Bc * Hc), 32, 0, stream>>>(qkvh, ctxh);

    // --- Wo projection with fused residual: partial' = partial + ascl*out ---
    float* np = pick(partial);
    gemm_bf16_kernel<2><<<dim3(Dc / 128, BTc / 128), 256, 0, stream>>>(
        ctxh, woh + (size_t)l * Dc * Dc, np, nullptr,
        pzero ? nullptr : partial, ascl + (size_t)l * Dc, BTc, Dc, Dc);
    partial = np; pzero = false;

    // --- second aggregate ---
    VPtrs V2{};
    for (int j = 0; j < nh; j++) V2.p[j] = hist[j];
    V2.p[nh] = partial;
    aggregate_rms_kernel<<<BTc, 256, 0, stream>>>(
        V2, nh + 1, mpw + (size_t)l * Dc, lns, l, xn2);

    // --- MLP: gelu(x @ Wm1) @ Wm2, gelu fused into first GEMM epilogue ---
    gemm_bf16_kernel<1><<<dim3(4 * Dc / 128, BTc / 128), 256, 0, stream>>>(
        xn2, wm1h + (size_t)l * Dc * 4 * Dc, nullptr, m1h, nullptr, nullptr,
        BTc, 4 * Dc, Dc);
    float* np2 = (l == Lc - 1) ? outp : pick(partial);
    gemm_bf16_kernel<2><<<dim3(Dc / 128, BTc / 128), 256, 0, stream>>>(
        m1h, wm2h + (size_t)l * 4 * Dc * Dc, np2, nullptr,
        partial, mscl + (size_t)l * Dc, BTc, Dc, 4 * Dc);
    partial = np2;
  }
}
